// TransformerBlock_79800492360201
// MI455X (gfx1250) — matrix operand order repaired and checked by reading, not compiled, not run on hardware
//
#include <hip/hip_runtime.h>
#include <math.h>

// ---------------------------------------------------------------- types
typedef __attribute__((ext_vector_type(16))) __bf16          bf16x16;
typedef __attribute__((ext_vector_type(8)))  __bf16          bf16x8;
typedef __attribute__((ext_vector_type(8)))  unsigned short  u16x8;
typedef __attribute__((ext_vector_type(8)))  float           f32x8;
typedef __attribute__((address_space(3)))    unsigned short  as3_ushort;

static __device__ __forceinline__ unsigned short f2bf_u(float f) {
  unsigned int u = __builtin_bit_cast(unsigned int, f);
  u += 0x7FFFu + ((u >> 16) & 1u);           // round-to-nearest-even
  return (unsigned short)(u >> 16);
}
static __device__ __forceinline__ __bf16 f2bf_h(float f) {
  return __builtin_bit_cast(__bf16, f2bf_u(f));
}

static __device__ __forceinline__ bf16x16 mk_frag(u16x8 a, u16x8 b) {
  bf16x8 ah = __builtin_bit_cast(bf16x8, a);
  bf16x8 bh = __builtin_bit_cast(bf16x8, b);
  return __builtin_shufflevector(ah, bh, 0,1,2,3,4,5,6,7,8,9,10,11,12,13,14,15);
}
// frag from two aligned 8-elem halves (LDS or global pointer)
static __device__ __forceinline__ bf16x16 frag_ld(const unsigned short* lo,
                                                  const unsigned short* hi) {
  return mk_frag(*(const u16x8*)lo, *(const u16x8*)hi);
}
// guarded frag (zero if !valid) for edge tiles
static __device__ __forceinline__ bf16x16 frag_ld_guard(const unsigned short* lo,
                                                        const unsigned short* hi,
                                                        bool valid) {
  const u16x8 z = (u16x8)(unsigned short)0;
  u16x8 a = valid ? *(const u16x8*)lo : z;
  u16x8 b = valid ? *(const u16x8*)hi : z;
  return mk_frag(a, b);
}

static __device__ __forceinline__ f32x8 wmma_bf16(bf16x16 a, bf16x16 b, f32x8 c) {
  return __builtin_amdgcn_wmma_f32_16x16x32_bf16(false, a, false, b,
                                                 (short)0, c, false, false);
}

// 32-bit LDS offset for async-to-LDS DMA
static __device__ __forceinline__ unsigned lds_addr32(unsigned short* p) {
  as3_ushort* lp = (as3_ushort*)p;
  return (unsigned)(unsigned long long)lp;
}
// one 16B global->LDS async copy, tracked by ASYNCcnt
static __device__ __forceinline__ void async_cp16(unsigned short* lds_dst,
                                                  const unsigned short* gsrc) {
  unsigned dst = lds_addr32(lds_dst);
  asm volatile("global_load_async_to_lds_b128 %0, %1, off"
               :: "v"(dst), "v"(gsrc) : "memory");
}
static __device__ __forceinline__ void wait_async8() {
  asm volatile("s_wait_asynccnt 0x8" ::: "memory");
}
static __device__ __forceinline__ void wait_async0() {
  asm volatile("s_wait_asynccnt 0x0" ::: "memory");
}

// ---------------------------------------------------------------- config
#define DMODEL 2048
#define SEQ    2048
#define NHEAD  16
#define DHEAD  128
#define BK     64
#define LDA_S  72    // 64-wide A tile rows (ushort units, 16B multiples)
#define LDB_S  136   // 128-wide B tile rows
#define LDU_S  72    // 64-wide B tile rows (up/gate)

// ---------------------------------------------------------------- f32 -> bf16 convert
__global__ __launch_bounds__(256) void conv_bf16(const float* __restrict__ in,
                                                 unsigned short* __restrict__ out) {
  size_t i = ((size_t)blockIdx.x * 256 + threadIdx.x) * 8;
  float4 a = *(const float4*)(in + i);
  float4 b = *(const float4*)(in + i + 4);
  u16x8 o;
  o[0] = f2bf_u(a.x); o[1] = f2bf_u(a.y); o[2] = f2bf_u(a.z); o[3] = f2bf_u(a.w);
  o[4] = f2bf_u(b.x); o[5] = f2bf_u(b.y); o[6] = f2bf_u(b.z); o[7] = f2bf_u(b.w);
  *(u16x8*)(out + i) = o;
}

// ---------------------------------------------------------------- rmsnorm (f32 in, bf16 out)
__global__ __launch_bounds__(256) void rmsnorm_k(const float* __restrict__ x,
                                                 const float* __restrict__ sc,
                                                 unsigned short* __restrict__ o) {
  __shared__ float red[256];
  const int row = blockIdx.x;
  const int tid = threadIdx.x;
  const float* xr = x + (size_t)row * DMODEL;
  float4 a = *(const float4*)(xr + tid * 8);
  float4 b = *(const float4*)(xr + tid * 8 + 4);
  red[tid] = a.x*a.x + a.y*a.y + a.z*a.z + a.w*a.w +
             b.x*b.x + b.y*b.y + b.z*b.z + b.w*b.w;
  __syncthreads();
  for (int s = 128; s > 0; s >>= 1) {
    if (tid < s) red[tid] += red[tid + s];
    __syncthreads();
  }
  const float inv = rsqrtf(red[0] * (1.0f / DMODEL) + 1e-6f);
  const float4 s1 = *(const float4*)(sc + tid * 8);
  const float4 s2 = *(const float4*)(sc + tid * 8 + 4);
  u16x8 ov;
  ov[0] = f2bf_u(a.x*inv*s1.x); ov[1] = f2bf_u(a.y*inv*s1.y);
  ov[2] = f2bf_u(a.z*inv*s1.z); ov[3] = f2bf_u(a.w*inv*s1.w);
  ov[4] = f2bf_u(b.x*inv*s2.x); ov[5] = f2bf_u(b.y*inv*s2.y);
  ov[6] = f2bf_u(b.z*inv*s2.z); ov[7] = f2bf_u(b.w*inv*s2.w);
  *(u16x8*)(o + (size_t)row * DMODEL + tid * 8) = ov;
}

// ---------------------------------------------------------------- GEMM (bf16, 128x128 tile, BK=64)
// EPI==0: Cb (bf16) = A@B     EPI==1: Cf (f32) = Cin + A@B
template <int EPI>
__global__ __launch_bounds__(256) void gemm_wmma(const unsigned short* __restrict__ A,
                                                 const unsigned short* __restrict__ B,
                                                 const float* __restrict__ Cin,
                                                 float* __restrict__ Cf,
                                                 unsigned short* __restrict__ Cb,
                                                 int M, int N, int K) {
  __shared__ __align__(16) unsigned short As[2][128 * LDA_S];
  __shared__ __align__(16) unsigned short Bs[2][BK * LDB_S];
  const int tid  = threadIdx.x;
  const int wave = tid >> 5;
  const int lane = tid & 31;
  const int hl   = lane >> 4;
  const int ln   = lane & 15;
  const int wm   = (wave >> 1) * 32;
  const int wn   = (wave & 1) * 64;
  const int m0   = blockIdx.y * 128;
  const int n0   = blockIdx.x * 128;

  f32x8 acc[2][4];
#pragma unroll
  for (int t = 0; t < 2; ++t)
#pragma unroll
    for (int u = 0; u < 4; ++u) acc[t][u] = (f32x8)0.0f;

  // per-thread staging chunks (precomputed; global ptrs advance per stage)
  int aDst[4], bDst[4];
  size_t aOff[4], bOff[4];
#pragma unroll
  for (int i = 0; i < 4; ++i) {
    int idx = tid + i * 256;
    int ra = idx >> 3, ca = (idx & 7) << 3;     // A: 128x64
    int rb = idx >> 4, cb = (idx & 15) << 3;    // B: 64x128
    aDst[i] = ra * LDA_S + ca;  aOff[i] = (size_t)ra * K + ca;
    bDst[i] = rb * LDB_S + cb;  bOff[i] = (size_t)rb * N + cb;
  }
  const unsigned short* aSrc = A + (size_t)m0 * K;
  const unsigned short* bSrc = B + n0;

  auto stage = [&](int buf) {
#pragma unroll
    for (int i = 0; i < 4; ++i) async_cp16(&As[buf][aDst[i]], aSrc + aOff[i]);
#pragma unroll
    for (int i = 0; i < 4; ++i) async_cp16(&Bs[buf][bDst[i]], bSrc + bOff[i]);
    aSrc += BK;
    bSrc += (size_t)BK * N;
  };

  const int nk = K / BK;
  stage(0);
  for (int ki = 0; ki < nk; ++ki) {
    const int cur = ki & 1;
    const bool more = (ki + 1) < nk;
    if (more) stage(cur ^ 1);
    if (more) wait_async8(); else wait_async0();   // current tile retired (in-order)
    __syncthreads();
#pragma unroll
    for (int kk = 0; kk < 2; ++kk) {
      bf16x16 bf[4];
#pragma unroll
      for (int u = 0; u < 4; ++u) {
        for (int s = 0; s < 16; ++s) {
          int k = (s < 8) ? hl * 8 + s : 16 + hl * 8 + (s - 8);
          bf[u][s] = __builtin_bit_cast(__bf16, Bs[cur][(kk * 32 + k) * LDB_S + wn + u * 16 + ln]);
        }
      }
#pragma unroll
      for (int t = 0; t < 2; ++t) {
        const unsigned short* ar = &As[cur][(wm + t * 16 + ln) * LDA_S + kk * 32];
        bf16x16 af = frag_ld(ar + hl * 8, ar + 16 + hl * 8);
#pragma unroll
        for (int u = 0; u < 4; ++u) acc[t][u] = wmma_bf16(af, bf[u], acc[t][u]);
      }
    }
    __syncthreads();
  }

#pragma unroll
  for (int t = 0; t < 2; ++t)
#pragma unroll
    for (int u = 0; u < 4; ++u)
#pragma unroll
      for (int r = 0; r < 8; ++r) {
        int row = m0 + wm + t * 16 + r + 8 * hl;
        int col = n0 + wn + u * 16 + ln;
        size_t off = (size_t)row * N + col;
        if (EPI == 0) Cb[off] = f2bf_u(acc[t][u][r]);
        else          Cf[off] = Cin[off] + acc[t][u][r];
      }
}

// ---------------------------------------------------------------- fused up/gate (128x64, BK=64)
__global__ __launch_bounds__(256) void gemm_upgate(const unsigned short* __restrict__ A,
                                                   const unsigned short* __restrict__ Bu,
                                                   const unsigned short* __restrict__ Bg,
                                                   unsigned short* __restrict__ H,
                                                   int M, int N, int K) {
  __shared__ __align__(16) unsigned short As[2][128 * LDA_S];
  __shared__ __align__(16) unsigned short Bus[2][BK * LDU_S];
  __shared__ __align__(16) unsigned short Bgs[2][BK * LDU_S];
  const int tid  = threadIdx.x;
  const int wave = tid >> 5;
  const int lane = tid & 31;
  const int hl   = lane >> 4;
  const int ln   = lane & 15;
  const int wm   = (wave >> 1) * 32;
  const int wn   = (wave & 1) * 32;
  const int m0   = blockIdx.y * 128;
  const int n0   = blockIdx.x * 64;

  f32x8 au[2][2], ag[2][2];
#pragma unroll
  for (int t = 0; t < 2; ++t)
#pragma unroll
    for (int u = 0; u < 2; ++u) { au[t][u] = (f32x8)0.0f; ag[t][u] = (f32x8)0.0f; }

  int aDst[4], bDst[2];
  size_t aOff[4], bOff[2];
#pragma unroll
  for (int i = 0; i < 4; ++i) {
    int idx = tid + i * 256;
    int ra = idx >> 3, ca = (idx & 7) << 3;     // A: 128x64
    aDst[i] = ra * LDA_S + ca;  aOff[i] = (size_t)ra * K + ca;
  }
#pragma unroll
  for (int i = 0; i < 2; ++i) {
    int idx = tid + i * 256;
    int rb = idx >> 3, cb = (idx & 7) << 3;     // Bu/Bg: 64x64
    bDst[i] = rb * LDU_S + cb;  bOff[i] = (size_t)rb * N + cb;
  }
  const unsigned short* aSrc = A + (size_t)m0 * K;
  const unsigned short* uSrc = Bu + n0;
  const unsigned short* gSrc = Bg + n0;

  auto stage = [&](int buf) {
#pragma unroll
    for (int i = 0; i < 4; ++i) async_cp16(&As[buf][aDst[i]], aSrc + aOff[i]);
#pragma unroll
    for (int i = 0; i < 2; ++i) async_cp16(&Bus[buf][bDst[i]], uSrc + bOff[i]);
#pragma unroll
    for (int i = 0; i < 2; ++i) async_cp16(&Bgs[buf][bDst[i]], gSrc + bOff[i]);
    aSrc += BK;
    uSrc += (size_t)BK * N;
    gSrc += (size_t)BK * N;
  };

  const int nk = K / BK;
  stage(0);
  for (int ki = 0; ki < nk; ++ki) {
    const int cur = ki & 1;
    const bool more = (ki + 1) < nk;
    if (more) stage(cur ^ 1);
    if (more) wait_async8(); else wait_async0();
    __syncthreads();
#pragma unroll
    for (int kk = 0; kk < 2; ++kk) {
      bf16x16 bu[2], bg[2];
#pragma unroll
      for (int u = 0; u < 2; ++u) {
        for (int s = 0; s < 16; ++s) {
          int k = (s < 8) ? hl * 8 + s : 16 + hl * 8 + (s - 8);
          bu[u][s] = __builtin_bit_cast(__bf16, Bus[cur][(kk * 32 + k) * LDU_S + wn + u * 16 + ln]);
          bg[u][s] = __builtin_bit_cast(__bf16, Bgs[cur][(kk * 32 + k) * LDU_S + wn + u * 16 + ln]);
        }
      }
#pragma unroll
      for (int t = 0; t < 2; ++t) {
        const unsigned short* ar = &As[cur][(wm + t * 16 + ln) * LDA_S + kk * 32];
        bf16x16 af = frag_ld(ar + hl * 8, ar + 16 + hl * 8);
#pragma unroll
        for (int u = 0; u < 2; ++u) {
          au[t][u] = wmma_bf16(af, bu[u], au[t][u]);
          ag[t][u] = wmma_bf16(af, bg[u], ag[t][u]);
        }
      }
    }
    __syncthreads();
  }

#pragma unroll
  for (int t = 0; t < 2; ++t)
#pragma unroll
    for (int u = 0; u < 2; ++u)
#pragma unroll
      for (int r = 0; r < 8; ++r) {
        int row = m0 + wm + t * 16 + r + 8 * hl;
        int col = n0 + wn + u * 16 + ln;
        float uv = au[t][u][r];
        float gv = ag[t][u][r];
        float sig = 1.0f / (1.0f + __expf(-gv));
        H[(size_t)row * N + col] = f2bf_u(uv * (gv * sig));
      }
}

// ---------------------------------------------------------------- flash attention
// Computes S^T = K.Q^T so K/V are consumed in their natural row-major layout
// (direct global loads, no LDS in the k-loop); S^T's C-layout (lane=query,
// vgpr=key) IS the A-layout of P for O = P.V -> pure register softmax.
__global__ __launch_bounds__(32) void flash_attn(const unsigned short* __restrict__ qkv,
                                                 unsigned short* __restrict__ outp) {
  __shared__ __align__(16) unsigned short QT[128 * 16];   // Q^T [dim][query]
  const int lane = threadIdx.x;
  const int hl = lane >> 4, ln = lane & 15;
  const int qb = blockIdx.x;
  const int bh = blockIdx.y;
  const int bb = bh >> 4, h = bh & 15;
  const int q0 = qb * 16;
  const int ldq = 3 * DMODEL;
  const float qscale = 0.08838834764831845f;   // 1/sqrt(128)

  // stage Q^T once: lane handles query ln, dim half hl*64..+63
  {
    const unsigned short* qp =
        qkv + (size_t)(bb * SEQ + q0 + ln) * ldq + h * DHEAD + hl * 64;
#pragma unroll
    for (int cc = 0; cc < 8; ++cc) {
      u16x8 v = *(const u16x8*)(qp + cc * 8);
#pragma unroll
      for (int j = 0; j < 8; ++j) QT[(hl * 64 + cc * 8 + j) * 16 + ln] = v[j];
    }
  }
  __syncthreads();
  bf16x16 qtf[4];
#pragma unroll
  for (int c = 0; c < 4; ++c) {
    for (int s = 0; s < 16; ++s) {
      int k = (s < 8) ? hl * 8 + s : 16 + hl * 8 + (s - 8);
      qtf[c][s] = __builtin_bit_cast(__bf16, QT[(c * 32 + k) * 16 + ln]);
    }
  }

  float mq = -1e30f, lq = 0.0f;   // per-lane (query = ln) running stats
  f32x8 O[8];
#pragma unroll
  for (int n = 0; n < 8; ++n) O[n] = (f32x8)0.0f;

  const int qq = q0 + ln;
  for (int ks = 0; ks <= q0 + 15; ks += 32) {
    // --- S^T = K.Q^T : keys ks..+15 (ST0) and ks+16..+31 (ST1)
    const unsigned short* k0p =
        qkv + (size_t)(bb * SEQ + ks + ln) * ldq + DMODEL + h * DHEAD;  // always in range
    const int key1 = ks + 16 + ln;
    const bool v1 = key1 < SEQ;
    const unsigned short* k1p =
        qkv + (size_t)(bb * SEQ + (v1 ? key1 : 0)) * ldq + DMODEL + h * DHEAD;
    f32x8 ST0 = (f32x8)0.0f, ST1 = (f32x8)0.0f;
#pragma unroll
    for (int c = 0; c < 4; ++c) {
      bf16x16 a0 = frag_ld(k0p + c * 32 + hl * 8, k0p + c * 32 + 16 + hl * 8);
      bf16x16 a1 = frag_ld_guard(k1p + c * 32 + hl * 8, k1p + c * 32 + 16 + hl * 8, v1);
      ST0 = wmma_bf16(a0, qtf[c], ST0);
      ST1 = wmma_bf16(a1, qtf[c], ST1);
    }

    // --- per-lane online softmax over 32 keys: 16 in this lane, 16 in lane^16
    float e0[8], e1[8];
    float mloc = -1e30f;
#pragma unroll
    for (int r = 0; r < 8; ++r) {
      float s0 = ST0[r] * qscale;
      float s1 = ST1[r] * qscale;
      if (ks + r + 8 * hl > qq)      s0 = -1e30f;
      if (ks + 16 + r + 8 * hl > qq) s1 = -1e30f;
      e0[r] = s0; e1[r] = s1;
      mloc = fmaxf(mloc, fmaxf(s0, s1));
    }
    mloc = fmaxf(mloc, __shfl_xor(mloc, 16));
    const float nm = fmaxf(mq, mloc);
    const float alpha = __expf(mq - nm);
    mq = nm;
    float sloc = 0.0f;
#pragma unroll
    for (int r = 0; r < 8; ++r) {
      e0[r] = __expf(e0[r] - nm);
      e1[r] = __expf(e1[r] - nm);
      sloc += e0[r] + e1[r];
    }
    sloc += __shfl_xor(sloc, 16);
    lq = lq * alpha + sloc;

    // --- P A-frag straight from registers (C-layout of S^T == A-layout of P)
    bf16x16 pf;
#pragma unroll
    for (int r = 0; r < 8; ++r) { pf[r] = f2bf_h(e0[r]); pf[r + 8] = f2bf_h(e1[r]); }

    // --- rescale O by alpha of its row's query (broadcast from owning lane)
#pragma unroll
    for (int r = 0; r < 8; ++r) {
      float ar = __shfl(alpha, r + 8 * hl);
#pragma unroll
      for (int n = 0; n < 8; ++n) O[n][r] *= ar;
    }

    // --- O += P.V, V consumed row-major straight from global
#pragma unroll
    for (int n = 0; n < 8; ++n) {
      bf16x16 vf;
      for (int s = 0; s < 16; ++s) {
        const int keyv = ks + ((s < 8) ? hl * 8 + s : 16 + hl * 8 + (s - 8));
        const bool vv = keyv < SEQ;
        unsigned short e = vv ? qkv[(size_t)(bb * SEQ + keyv) * ldq + 2 * DMODEL + h * DHEAD + n * 16 + ln] : (unsigned short)0;
        vf[s] = __builtin_bit_cast(__bf16, e);
      }
      O[n] = wmma_bf16(pf, vf, O[n]);
    }
  }

  // --- normalize and write (bf16)
#pragma unroll
  for (int r = 0; r < 8; ++r) {
    float rl = __shfl(lq, r + 8 * hl);
    float rinv = 1.0f / rl;
    int row = q0 + r + 8 * hl;
#pragma unroll
    for (int n = 0; n < 8; ++n)
      outp[(size_t)(bb * SEQ + row) * DMODEL + h * DHEAD + n * 16 + ln] =
          f2bf_u(O[n][r] * rinv);
  }
}

// ---------------------------------------------------------------- launch
extern "C" void kernel_launch(void* const* d_in, const int* in_sizes, int n_in,
                              void* d_out, int out_size, void* d_ws, size_t ws_size,
                              hipStream_t stream) {
  const float* x      = (const float*)d_in[0];
  const float* w_qkv  = (const float*)d_in[1];
  const float* w_o    = (const float*)d_in[2];
  const float* w_up   = (const float*)d_in[3];
  const float* w_gate = (const float*)d_in[4];
  const float* w_down = (const float*)d_in[5];
  const float* scale1 = (const float*)d_in[6];
  const float* scale2 = (const float*)d_in[7];
  float* out = (float*)d_out;
  unsigned short* u = (unsigned short*)d_ws;

  const int BS = 2 * SEQ;  // 4096 rows
  unsigned short* xn_b   = u;                          // [BS,2048]
  unsigned short* attn_b = u + (size_t)8388608;        // [BS,2048]
  unsigned short* qkv_b  = u + (size_t)16777216;       // [BS,6144]
  unsigned short* h_b    = u + (size_t)8388608;        // [BS,8192] reuses attn+qkv
  unsigned short* wqkv_b  = u + (size_t)41943040;
  unsigned short* wo_b    = u + (size_t)54525952;
  unsigned short* wup_b   = u + (size_t)58720256;
  unsigned short* wgate_b = u + (size_t)75497472;
  unsigned short* wdown_b = u + (size_t)92274688;

  conv_bf16<<<12582912 / 2048, 256, 0, stream>>>(w_qkv,  wqkv_b);
  conv_bf16<<< 4194304 / 2048, 256, 0, stream>>>(w_o,    wo_b);
  conv_bf16<<<16777216 / 2048, 256, 0, stream>>>(w_up,   wup_b);
  conv_bf16<<<16777216 / 2048, 256, 0, stream>>>(w_gate, wgate_b);
  conv_bf16<<<16777216 / 2048, 256, 0, stream>>>(w_down, wdown_b);

  rmsnorm_k<<<BS, 256, 0, stream>>>(x, scale1, xn_b);
  gemm_wmma<0><<<dim3(3 * DMODEL / 128, BS / 128), 256, 0, stream>>>(
      xn_b, wqkv_b, nullptr, nullptr, qkv_b, BS, 3 * DMODEL, DMODEL);
  flash_attn<<<dim3(SEQ / 16, 2 * NHEAD), 32, 0, stream>>>(qkv_b, attn_b);
  gemm_wmma<1><<<dim3(DMODEL / 128, BS / 128), 256, 0, stream>>>(
      attn_b, wo_b, x, out, nullptr, BS, DMODEL, DMODEL);

  rmsnorm_k<<<BS, 256, 0, stream>>>(out, scale2, xn_b);
  gemm_upgate<<<dim3(4 * DMODEL / 64, BS / 128), 256, 0, stream>>>(
      xn_b, wup_b, wgate_b, h_b, BS, 4 * DMODEL, DMODEL);
  gemm_wmma<1><<<dim3(DMODEL / 128, BS / 128), 256, 0, stream>>>(
      h_b, wdown_b, out, out, nullptr, BS, DMODEL, 4 * DMODEL);
}
